// MIL_17051020165232
// MI455X (gfx1250) — matrix operand order repaired and checked by reading, not compiled, not run on hardware
//
#include <hip/hip_runtime.h>
#include <math.h>

// Problem constants (from reference)
#define BATCH   32
#define TLEN    2048
#define DDIM    1024
#define H1DIM   512
#define NTOT    (BATCH * TLEN)   // 65536 rows
#define NEGV    (-1e30f)

// Main GEMM kernel tiling
#define TPB1     512             // 16 wave32 waves
#define ROWS_WG  64              // rows per workgroup
#define KCH      32              // K chunk (matches wmma 16x16x32)
#define NKT      (DDIM / KCH)    // 32 K-chunks
#define SA_PITCH 40              // padded pitch (bf16 elems): 80B -> conflict-free b128 LDS reads
#define SB_PITCH 40

typedef __attribute__((ext_vector_type(16))) __bf16 bf16x16;
typedef __attribute__((ext_vector_type(8)))  __bf16 bf16x8;
typedef __attribute__((ext_vector_type(2)))  __bf16 bf16x2;
typedef __attribute__((ext_vector_type(8)))  float  f32x8;

union BF16Frag { bf16x16 v; bf16x8 h[2]; };

// native fp32 -> bf16 convert (RNE), packed into one dword (v_cvt_pk_bf16_f32)
static __device__ __forceinline__ unsigned int pack_bf2(float lo, float hi) {
  bf16x2 t;
  t[0] = (__bf16)lo;
  t[1] = (__bf16)hi;
  return __builtin_bit_cast(unsigned int, t);
}

// ---------------------------------------------------------------------------
// Kernel 0: fold layers 2+3 (linear): w[h] = sum_k W2[h,k]*W3[k],
//           c = sum_k b2[k]*W3[k] + b3.
// ---------------------------------------------------------------------------
__global__ void MIL_prep_kernel(const float* __restrict__ W2,
                                const float* __restrict__ b2,
                                const float* __restrict__ W3,
                                const float* __restrict__ b3,
                                float* __restrict__ wvec,
                                float* __restrict__ cptr) {
  int h = threadIdx.x;
  if (h < H1DIM) {
    float s = 0.f;
#pragma unroll
    for (int j = 0; j < 32; ++j) s += W2[h * 32 + j] * W3[j];
    wvec[h] = s;
  }
  if (h == 0) {
    float c = b3[0];
#pragma unroll
    for (int j = 0; j < 32; ++j) c += b2[j] * W3[j];
    cptr[0] = c;
  }
}

// ---------------------------------------------------------------------------
// Kernel 1: fused  logits = sigmoid( relu(A@W1 + b1) . w + c )
// 64 rows x 512 cols per block; double-buffered LDS; bf16 WMMA, f32 accum.
// Phase order per K-chunk (enforced with sched_barrier):
//   [issue next-chunk global loads] | [frag loads + 8 wmma] | [cvt + LDS store]
// ---------------------------------------------------------------------------
__global__ __launch_bounds__(TPB1) void MIL_gemm_logits_kernel(
    const float* __restrict__ A,     // [NTOT, DDIM]
    const float* __restrict__ W1,    // [DDIM, H1DIM]
    const float* __restrict__ b1,    // [H1DIM]
    const float* __restrict__ wvec,  // [H1DIM]
    const float* __restrict__ cptr,  // [1]
    float* __restrict__ logits) {    // [NTOT]
  __shared__ unsigned short sA[2][ROWS_WG * SA_PITCH];  // 64x32 bf16 (ping-pong)
  __shared__ unsigned short sB[2][H1DIM * SB_PITCH];    // 512x32 bf16 W1^T chunk (ping-pong)
  __shared__ float rowsum[ROWS_WG];

  const int tid  = threadIdx.x;
  const int lane = tid & 31;
  const int wv   = tid >> 5;       // 0..15
  const int rt   = wv >> 2;        // row tile 0..3
  const int cg   = wv & 3;         // col group 0..3 (128 cols each)
  const int n    = lane & 15;
  const int half = lane >> 4;
  const int row0 = blockIdx.x * ROWS_WG;

  // staging coordinates (fixed per thread)
  const int sa_r  = tid >> 3;            // 0..63
  const int sa_c4 = (tid & 7) << 2;      // 0,4,..28
  const float* aPtr = A + (size_t)(row0 + sa_r) * DDIM + sa_c4;

  if (tid < ROWS_WG) rowsum[tid] = 0.f;

  f32x8 acc[8];
#pragma unroll
  for (int ct = 0; ct < 8; ++ct) acc[ct] = {};

  float4 aR;      // staged A chunk (1 float4/thread)
  float4 bR[8];   // staged W1 chunk (4 kk-pairs x 2 rows)

  // ---- prologue: load + stage chunk 0 into buffer 0 ----
  aR = *(const float4*)(aPtr);
#pragma unroll
  for (int j = 0; j < 4; ++j) {
    const int p  = tid + j * TPB1;
    const int kk = (p >> 7) << 1;          // even kk
    const int c4 = (p & 127) << 2;
    bR[2 * j]     = *(const float4*)(W1 + (size_t)(kk)     * H1DIM + c4);
    bR[2 * j + 1] = *(const float4*)(W1 + (size_t)(kk + 1) * H1DIM + c4);
  }
  {
    unsigned int* d = (unsigned int*)&sA[0][sa_r * SA_PITCH + sa_c4];
    d[0] = pack_bf2(aR.x, aR.y);
    d[1] = pack_bf2(aR.z, aR.w);
#pragma unroll
    for (int j = 0; j < 4; ++j) {
      const int p  = tid + j * TPB1;
      const int kk = (p >> 7) << 1;
      const int c4 = (p & 127) << 2;
      const float* r0 = (const float*)&bR[2 * j];
      const float* r1 = (const float*)&bR[2 * j + 1];
#pragma unroll
      for (int i = 0; i < 4; ++i)
        *(unsigned int*)&sB[0][(c4 + i) * SB_PITCH + kk] = pack_bf2(r0[i], r1[i]);
    }
  }
  __syncthreads();

  for (int kt = 0; kt < NKT; ++kt) {
    const int buf = kt & 1;
    const int k1  = (kt + 1) * KCH;

    // ==== phase 1: issue ALL next-chunk global loads (no waits here) ====
    if (kt + 1 < NKT) {
      aR = *(const float4*)(aPtr + k1);
#pragma unroll
      for (int j = 0; j < 4; ++j) {
        const int p  = tid + j * TPB1;
        const int kk = (p >> 7) << 1;
        const int c4 = (p & 127) << 2;
        bR[2 * j]     = *(const float4*)(W1 + (size_t)(k1 + kk)     * H1DIM + c4);
        bR[2 * j + 1] = *(const float4*)(W1 + (size_t)(k1 + kk + 1) * H1DIM + c4);
      }
    }
    __builtin_amdgcn_sched_barrier(0);  // pin loads above the compute phase

    // ==== phase 2: compute on current buffer ====
    const unsigned short* cA = sA[buf];
    const unsigned short* cB = sB[buf];
    BF16Frag af;
    {
      const int r = rt * 16 + n;
      af.h[0] = *(const bf16x8*)&cA[r * SA_PITCH + 8 * half];       // K 0..7  (+8*half)
      af.h[1] = *(const bf16x8*)&cA[r * SA_PITCH + 16 + 8 * half];  // K 16..23(+8*half)
    }
    BF16Frag bfr[8];
#pragma unroll
    for (int ct = 0; ct < 8; ++ct) {
      const int col = cg * 128 + ct * 16 + n;
      bfr[ct].h[0] = *(const bf16x8*)&cB[col * SB_PITCH + 8 * half];
      bfr[ct].h[1] = *(const bf16x8*)&cB[col * SB_PITCH + 16 + 8 * half];
    }
#pragma unroll
    for (int ct = 0; ct < 8; ++ct) {
      acc[ct] = __builtin_amdgcn_wmma_f32_16x16x32_bf16(
          false, af.v, false, bfr[ct].v, (short)0, acc[ct], false, false);
    }
    __builtin_amdgcn_sched_barrier(0);  // keep cvt/stores (and their waits) below

    // ==== phase 3: convert + stage next chunk into the other buffer ====
    if (kt + 1 < NKT) {
      const int obuf = buf ^ 1;
      unsigned int* d = (unsigned int*)&sA[obuf][sa_r * SA_PITCH + sa_c4];
      d[0] = pack_bf2(aR.x, aR.y);
      d[1] = pack_bf2(aR.z, aR.w);
#pragma unroll
      for (int j = 0; j < 4; ++j) {
        const int p  = tid + j * TPB1;
        const int kk = (p >> 7) << 1;
        const int c4 = (p & 127) << 2;
        const float* r0 = (const float*)&bR[2 * j];
        const float* r1 = (const float*)&bR[2 * j + 1];
#pragma unroll
        for (int i = 0; i < 4; ++i)
          *(unsigned int*)&sB[obuf][(c4 + i) * SB_PITCH + kk] = pack_bf2(r0[i], r1[i]);
      }
    }
    __syncthreads();
  }

  // ---- fused epilogue: relu(acc + b1[col]) * w[col], reduce over columns ----
  float partial[8];
#pragma unroll
  for (int v = 0; v < 8; ++v) partial[v] = 0.f;
#pragma unroll
  for (int ct = 0; ct < 8; ++ct) {
    const int col  = cg * 128 + ct * 16 + n;
    const float bb = b1[col];
    const float ww = wvec[col];
#pragma unroll
    for (int v = 0; v < 8; ++v) {
      // C/D layout: lanes 0-15 vgpr v -> M=v; lanes 16-31 -> M=v+8; N=lane&15
      float val = acc[ct][v] + bb;
      partial[v] += fmaxf(val, 0.f) * ww;
    }
  }
#pragma unroll
  for (int m = 1; m < 16; m <<= 1) {  // reduce across 16 N-lanes (stays in half)
#pragma unroll
    for (int v = 0; v < 8; ++v) partial[v] += __shfl_xor(partial[v], m, 32);
  }
  if (n == 0) {  // lanes 0 and 16 hold full sums for their 8 rows
#pragma unroll
    for (int v = 0; v < 8; ++v)
      atomicAdd(&rowsum[rt * 16 + 8 * half + v], partial[v]);
  }
  __syncthreads();
  if (tid < ROWS_WG) {
    const float x = rowsum[tid] + cptr[0];
    logits[row0 + tid] = 1.0f / (1.0f + __expf(-x));
  }
}

// ---------------------------------------------------------------------------
// Kernel 2: per-row masked top-k mean via bitonic sort (T=2048 fits LDS).
// ---------------------------------------------------------------------------
__global__ void MIL_topk_kernel(const float* __restrict__ logits,
                                const int* __restrict__ seq_len,
                                float* __restrict__ out) {
  __shared__ float s[TLEN];
  const int b   = blockIdx.x;
  const int tid = threadIdx.x;
  const int L   = seq_len[b];
  for (int i = tid; i < TLEN; i += blockDim.x)
    s[i] = (i < L) ? logits[b * TLEN + i] : NEGV;
  __syncthreads();
  for (int kk = 2; kk <= TLEN; kk <<= 1) {
    for (int j = kk >> 1; j > 0; j >>= 1) {
      for (int i = tid; i < TLEN; i += blockDim.x) {
        const int ixj = i ^ j;
        if (ixj > i) {
          const float a = s[i], c = s[ixj];
          const bool desc = ((i & kk) == 0);
          if (desc ? (a < c) : (a > c)) { s[i] = c; s[ixj] = a; }
        }
      }
      __syncthreads();
    }
  }
  if (tid == 0) {
    const int k = L / 16 + 1;  // k <= 129, guaranteed k <= L
    float sum = 0.f;
    for (int i = 0; i < k; ++i) sum += s[i];
    out[b] = sum / (float)k;
  }
}

// ---------------------------------------------------------------------------
extern "C" void kernel_launch(void* const* d_in, const int* in_sizes, int n_in,
                              void* d_out, int out_size, void* d_ws, size_t ws_size,
                              hipStream_t stream) {
  const float* avf     = (const float*)d_in[0];  // [B,T,D]
  const int*   seq_len = (const int*)d_in[1];    // [B]
  const float* W1      = (const float*)d_in[2];  // [D,H1]
  const float* b1      = (const float*)d_in[3];  // [H1]
  const float* W2      = (const float*)d_in[4];  // [H1,H2]
  const float* b2      = (const float*)d_in[5];  // [H2]
  const float* W3      = (const float*)d_in[6];  // [H2,1]
  const float* b3      = (const float*)d_in[7];  // [1]

  // workspace layout (floats): [0..511]=w, [512]=c, [1024..1024+65535]=logits
  float* wvec   = (float*)d_ws;
  float* cptr   = wvec + 512;
  float* logits = wvec + 1024;
  float* out    = (float*)d_out;

  MIL_prep_kernel<<<1, 512, 0, stream>>>(W2, b2, W3, b3, wvec, cptr);
  MIL_gemm_logits_kernel<<<NTOT / ROWS_WG, TPB1, 0, stream>>>(avf, W1, b1, wvec, cptr, logits);
  MIL_topk_kernel<<<BATCH, 256, 0, stream>>>(logits, seq_len, out);
}
